// NumFeaturesEncoder_20177756356729
// MI455X (gfx1250) — matrix operand order repaired and checked by reading, not compiled, not run on hardware
//
#include <hip/hip_runtime.h>

// ---------------------------------------------------------------------------
// KAN forward for MI455X (gfx1250, wave32, WMMA + async global->LDS).
// Each kan_linear(x) = gelu(x) @ base_w^T + einsum('bik,oik->bo', B(x), spline_w)
// is reformulated as ONE bf16 GEMM with K = fin*12:
//   A[b][i*12+0]  = gelu(x[b][i])          W[o][i*12+0]  = base_w[o][i]
//   A[b][i*12+1+s]= bspline_s(x[b][i])     W[o][i*12+1+s]= spline_w[o][i][s]
// The expanded A panels (50-100 MB bf16) fit in the 192 MB L2, so we
// materialize them once per layer and run a v_wmma_f32_16x16x32_bf16 GEMM.
// GEMM tile staging uses GLOBAL_LOAD_ASYNC_TO_LDS_B128 (ASYNCcnt) so the
// double-buffer fill bypasses VGPRs entirely.
// ---------------------------------------------------------------------------

typedef __attribute__((ext_vector_type(16))) __bf16 v16bf;
typedef __attribute__((ext_vector_type(8)))  __bf16 v8bf;
typedef __attribute__((ext_vector_type(8)))  float  v8f;

#define KAN_BATCH 8192

// ---- float -> bf16, round-to-nearest-even --------------------------------
__device__ __forceinline__ unsigned short f2bf(float f) {
  union { float f; unsigned u; } c; c.f = f;
  unsigned u = c.u;
  return (unsigned short)((u + 0x7FFFu + ((u >> 16) & 1u)) >> 16);
}

// store 12 bf16 (24 B, 8 B aligned) as three 8-byte writes
__device__ __forceinline__ void store12(__bf16* dst, const float* v) {
  unsigned u[6];
#pragma unroll
  for (int i = 0; i < 6; ++i)
    u[i] = (unsigned)f2bf(v[2 * i]) | ((unsigned)f2bf(v[2 * i + 1]) << 16);
  uint2* p = reinterpret_cast<uint2*>(dst);
  p[0] = make_uint2(u[0], u[1]);
  p[1] = make_uint2(u[2], u[3]);
  p[2] = make_uint2(u[4], u[5]);
}

// ---- CDNA5 async copy: 16 B global -> LDS, tracked by ASYNCcnt ------------
// GVS mode: saddr = uniform 64-bit base (SGPR pair), vaddr = per-lane I32
// byte offset. LDS destination byte offset = low 32 bits of the generic
// pointer (ISA 10.2: LDS aperture truncates addr[31:0]).
__device__ __forceinline__ void async_b128(void* lds_ptr, const void* gbase,
                                           unsigned gbyte_off) {
  unsigned lds_off = (unsigned)(uintptr_t)lds_ptr;
  asm volatile("global_load_async_to_lds_b128 %0, %1, %2"
               :
               : "v"(lds_off), "v"(gbyte_off), "s"(gbase)
               : "memory");
}

__device__ __forceinline__ void wait_async0() {
  asm volatile("s_wait_asynccnt 0" ::: "memory");
}

// ---------------------------------------------------------------------------
// Pack base_w [FOUT x FIN] and spline_w [FOUT x FIN x 11] into
// Wc [FOUT x FIN*12] bf16 (row-major, contiguous K per output row).
// ---------------------------------------------------------------------------
__global__ __launch_bounds__(256) void kan_pack_w(const float* __restrict__ base_w,
                                                  const float* __restrict__ spline_w,
                                                  __bf16* __restrict__ Wc, int total) {
  int idx = blockIdx.x * 256 + threadIdx.x;  // idx = o*FIN + i
  if (idx >= total) return;
  float o[12];
  o[0] = base_w[idx];
#pragma unroll
  for (int s = 0; s < 11; ++s) o[1 + s] = spline_w[(size_t)idx * 11 + s];
  store12(Wc + (size_t)idx * 12, o);
}

// ---------------------------------------------------------------------------
// Expand activations: per scalar x -> [gelu(x), B0(x)..B10(x)] in bf16.
// Uniform knots t_j = -1.75 + 0.25*j (j=0..14), Cox-de Boor, order 3.
// ---------------------------------------------------------------------------
template <int FIN>
__global__ __launch_bounds__(256) void kan_expand(const float* __restrict__ x,
                                                  __bf16* __restrict__ A) {
  int idx = blockIdx.x * 256 + threadIdx.x;
  if (idx >= KAN_BATCH * FIN) return;
  float v = x[idx];

  float o[12];
  o[0] = 0.5f * v * (1.0f + erff(v * 0.70710678118654752f));  // exact gelu

  float b[14];
#pragma unroll
  for (int j = 0; j < 14; ++j) {
    float t0 = -1.75f + 0.25f * (float)j;
    b[j] = (v >= t0 && v < t0 + 0.25f) ? 1.0f : 0.0f;
  }
#pragma unroll
  for (int k = 1; k <= 3; ++k) {
    float inv = 1.0f / (0.25f * (float)k);  // uniform-knot denominator
#pragma unroll
    for (int j = 0; j < 14 - k; ++j) {
      float tj = -1.75f + 0.25f * (float)j;
      float lf = (v - tj) * inv;
      float rf = ((tj + 0.25f * (float)(k + 1)) - v) * inv;
      b[j] = lf * b[j] + rf * b[j + 1];
    }
  }
#pragma unroll
  for (int j = 0; j < 11; ++j) o[1 + j] = b[j];

  store12(A + (size_t)idx * 12, o);
}

// ---------------------------------------------------------------------------
// bf16 WMMA GEMM: out[BATCH x FOUT] = A[BATCH x K] * Wc[FOUT x K]^T, K=FIN*12.
// Tile: BM=128, BN=64, BK=64. 256 threads = 8 wave32 in a 4x2 wave grid,
// each wave owns a 32x32 C tile (4 accumulators of v8f).
// Double-buffered LDS filled by async global->LDS DMA; rows padded to 72 bf16
// (144 B) -> conflict-free 64-bank pattern for the fragment reads.
// Fragment layouts per CDNA5 ISA 7.12.2:
//   A 16x32 bf16: lane<16 row=lane holds K kk+[0..7],kk+[16..23];
//                 lane>=16 same row set, K kk+[8..15],kk+[24..31].
//   B 32x16 bf16: lane<16 col=lane holds K kk+[0..7],kk+[16..23]; lane>=16 kk+[8..15],kk+[24..31].
// ---------------------------------------------------------------------------
template <int FOUT, int FIN, bool FUSE_BN>
__global__ __launch_bounds__(256) void kan_gemm(
    const __bf16* __restrict__ A, const __bf16* __restrict__ W,
    float* __restrict__ out,
    const float* __restrict__ bn_g, const float* __restrict__ bn_b,
    const float* __restrict__ bn_m, const float* __restrict__ bn_v) {
  constexpr int K   = FIN * 12;
  constexpr int BM  = 128;
  constexpr int BN_ = 64;
  constexpr int BK  = 64;
  constexpr int LD  = BK + 8;  // LDS row stride in bf16 (144 B)
  constexpr int KT  = K / BK;

  alignas(16) __shared__ __bf16 As[2][BM * LD];  // 36,864 B
  alignas(16) __shared__ __bf16 Bs[2][BN_ * LD]; // 18,432 B (total 55,296 B)

  const int tid  = threadIdx.x;
  const int wid  = tid >> 5;
  const int lane = tid & 31;
  const int half = lane >> 4;
  const int l16  = lane & 15;

  const int mBlock = blockIdx.x * BM;
  const int nBlock = blockIdx.y * BN_;
  const int wm = (wid & 3) * 32;   // wave row offset in tile
  const int wn = (wid >> 2) * 32;  // wave col offset in tile

  v8f acc[2][2] = {};

  auto loadTiles = [&](int kb, int buf) {
    // A tile: 128 rows x 64 bf16 = 1024 chunks of 16 B -> 4 per thread
#pragma unroll
    for (int i = 0; i < 4; ++i) {
      int c  = tid + i * 256;
      int r  = c >> 3;
      int k8 = (c & 7) << 3;
      async_b128(&As[buf][r * LD + k8], A,
                 (unsigned)(((unsigned)(mBlock + r) * (unsigned)K +
                             (unsigned)(kb + k8)) * 2u));
    }
    // B tile: 64 rows x 64 bf16 = 512 chunks -> 2 per thread
#pragma unroll
    for (int i = 0; i < 2; ++i) {
      int c  = tid + i * 256;
      int r  = c >> 3;
      int k8 = (c & 7) << 3;
      async_b128(&Bs[buf][r * LD + k8], W,
                 (unsigned)(((unsigned)(nBlock + r) * (unsigned)K +
                             (unsigned)(kb + k8)) * 2u));
    }
  };

  loadTiles(0, 0);
  wait_async0();
  __syncthreads();

  for (int kt = 0; kt < KT; ++kt) {
    const int buf = kt & 1;
    if (kt + 1 < KT) loadTiles((kt + 1) * BK, buf ^ 1);
    if (kt + 2 < KT) {  // warm GL2 two tiles ahead (global_prefetch_b8)
      int kb2 = (kt + 2) * BK;
      __builtin_prefetch(A + (size_t)(mBlock + (tid & 127)) * K + kb2, 0, 0);
      __builtin_prefetch(W + (size_t)(nBlock + (tid & 63)) * K + kb2, 0, 0);
    }

#pragma unroll
    for (int kk = 0; kk < BK; kk += 32) {
      v16bf afr[2], bfr[2];
#pragma unroll
      for (int mt = 0; mt < 2; ++mt) {
        const __bf16* p = &As[buf][(wm + mt * 16 + l16) * LD + kk + half * 8];
        v8bf a0 = *reinterpret_cast<const v8bf*>(p);
        v8bf a1 = *reinterpret_cast<const v8bf*>(p + 16);
        afr[mt] = __builtin_shufflevector(a0, a1, 0, 1, 2, 3, 4, 5, 6, 7, 8, 9,
                                          10, 11, 12, 13, 14, 15);
      }
#pragma unroll
      for (int nt = 0; nt < 2; ++nt) {
        const __bf16* p = &Bs[buf][(wn + nt * 16 + l16) * LD + kk + half * 8];
        v8bf b0 = *reinterpret_cast<const v8bf*>(p);
        v8bf b1 = *reinterpret_cast<const v8bf*>(p + 16);
        bfr[nt] = __builtin_shufflevector(b0, b1, 0, 1, 2, 3, 4, 5, 6, 7, 8, 9,
                                          10, 11, 12, 13, 14, 15);
      }
#pragma unroll
      for (int mt = 0; mt < 2; ++mt)
#pragma unroll
        for (int nt = 0; nt < 2; ++nt)
          acc[mt][nt] = __builtin_amdgcn_wmma_f32_16x16x32_bf16(
              false, afr[mt], false, bfr[nt], (short)0, acc[mt][nt], false,
              false);
    }
    wait_async0();     // own async fills of buf^1 complete
    __syncthreads();   // everyone's fills visible
  }

  // Epilogue. C layout (ISA 7.12.2): vgpr r, lane L -> row r + (L/16)*8, col L%16.
#pragma unroll
  for (int nt = 0; nt < 2; ++nt) {
    const int col = nBlock + wn + nt * 16 + l16;
    float g = 1.0f, bb = 0.0f, mu = 0.0f, inv = 1.0f;
    if constexpr (FUSE_BN) {
      g   = bn_g[col];
      bb  = bn_b[col];
      mu  = bn_m[col];
      inv = rsqrtf(bn_v[col] + 1e-5f);
    }
#pragma unroll
    for (int mt = 0; mt < 2; ++mt) {
#pragma unroll
      for (int r = 0; r < 8; ++r) {
        const int row = mBlock + wm + mt * 16 + half * 8 + r;
        float vv = acc[mt][nt][r];
        if constexpr (FUSE_BN) vv = g * (vv - mu) * inv + bb;
        out[(size_t)row * FOUT + col] = vv;
      }
    }
  }
}

// ---------------------------------------------------------------------------
// Launch: pack weights, then per layer expand -> GEMM. BN fused in last GEMM.
// Workspace usage ~146 MB (Wc panels + one 100 MB A panel + two activations).
// ---------------------------------------------------------------------------
extern "C" void kernel_launch(void* const* d_in, const int* in_sizes, int n_in,
                              void* d_out, int out_size, void* d_ws,
                              size_t ws_size, hipStream_t stream) {
  const float* x    = (const float*)d_in[0];
  const float* bw0  = (const float*)d_in[2];
  const float* sw0  = (const float*)d_in[3];
  const float* bw1  = (const float*)d_in[5];
  const float* sw1  = (const float*)d_in[6];
  const float* bw2  = (const float*)d_in[8];
  const float* sw2  = (const float*)d_in[9];
  const float* bn_g = (const float*)d_in[10];
  const float* bn_b = (const float*)d_in[11];
  const float* bn_m = (const float*)d_in[12];
  const float* bn_v = (const float*)d_in[13];

  char*  ws  = (char*)d_ws;
  size_t off = 0;
  auto alloc = [&](size_t bytes) {
    void* p = ws + off;
    off += (bytes + 255) & ~size_t(255);
    return p;
  };
  __bf16* Wc0  = (__bf16*)alloc((size_t)512 * 3072 * 2);
  __bf16* Wc1  = (__bf16*)alloc((size_t)512 * 6144 * 2);
  __bf16* Wc2  = (__bf16*)alloc((size_t)256 * 6144 * 2);
  __bf16* Abuf = (__bf16*)alloc((size_t)KAN_BATCH * 6144 * 2);
  float*  act1 = (float*)alloc((size_t)KAN_BATCH * 512 * 4);
  float*  act2 = (float*)alloc((size_t)KAN_BATCH * 512 * 4);

  kan_pack_w<<<(512 * 256 + 255) / 256, 256, 0, stream>>>(bw0, sw0, Wc0, 512 * 256);
  kan_pack_w<<<(512 * 512 + 255) / 256, 256, 0, stream>>>(bw1, sw1, Wc1, 512 * 512);
  kan_pack_w<<<(256 * 512 + 255) / 256, 256, 0, stream>>>(bw2, sw2, Wc2, 256 * 512);

  // Layer 0: 256 -> 512
  kan_expand<256><<<(KAN_BATCH * 256) / 256, 256, 0, stream>>>(x, Abuf);
  kan_gemm<512, 256, false><<<dim3(KAN_BATCH / 128, 512 / 64), 256, 0, stream>>>(
      Abuf, Wc0, act1, nullptr, nullptr, nullptr, nullptr);

  // Layer 1: 512 -> 512
  kan_expand<512><<<(KAN_BATCH * 512) / 256, 256, 0, stream>>>(act1, Abuf);
  kan_gemm<512, 512, false><<<dim3(KAN_BATCH / 128, 512 / 64), 256, 0, stream>>>(
      Abuf, Wc1, act2, nullptr, nullptr, nullptr, nullptr);

  // Layer 2: 512 -> 256, fused batchnorm epilogue
  kan_expand<512><<<(KAN_BATCH * 512) / 256, 256, 0, stream>>>(act2, Abuf);
  kan_gemm<256, 512, true><<<dim3(KAN_BATCH / 128, 256 / 64), 256, 0, stream>>>(
      Abuf, Wc2, (float*)d_out, bn_g, bn_b, bn_m, bn_v);
}
